// CompAggWikiQA_28278064676921
// MI455X (gfx1250) — matrix operand order repaired and checked by reading, not compiled, not run on hardware
//
#include <hip/hip_runtime.h>

// CompAgg WikiQA pipeline for MI455X (gfx1250), wave32, WMMA bf16 16x16x32.
// All B operands pre-packed into WMMA fragment layout -> aligned b128 loads,
// no predicated scalar loads in the GEMM K-loops.

typedef __bf16 bf16;
typedef __attribute__((ext_vector_type(16))) __bf16 v16bf;
typedef __attribute__((ext_vector_type(8)))  __bf16 v8bf;
typedef __attribute__((ext_vector_type(8)))  float  v8f;

constexpr int Tn = 65536;   // total answer tokens
constexpr int Mn = 256;     // hidden
constexpr int Qn = 64;      // query tokens
constexpr int Kn = 1024;    // answers
constexpr int En = 300;     // embedding dim (padded to 320 for the K-loop)

__device__ __forceinline__ v8f wmma_bf16(v16bf a, v16bf b, v8f c) {
  return __builtin_amdgcn_wmma_f32_16x16x32_bf16(false, a, false, b, (short)0, c,
                                                 false, false);
}

// A-matrix 16x32 bf16 fragment (ISA 7.12.2). No guards: callers guarantee
// in-range, 16B-aligned rows. Two aligned 16B vector loads.
__device__ __forceinline__ v16bf load_a_frag(const bf16* __restrict__ A, int lda,
                                             int row0, int k0) {
  int lane = threadIdx.x & 31;
  const bf16* rowp = A + (size_t)(row0 + (lane & 15)) * lda + k0 + ((lane >> 4) << 4);
  v8bf lo = *(const v8bf*)rowp;
  v8bf hi = *(const v8bf*)(rowp + 8);
  v16bf a;
#pragma unroll
  for (int j = 0; j < 8; ++j) { a[j] = lo[j]; a[j + 8] = hi[j]; }
  return a;
}

// B fragment from pre-packed layout: frag (kt,nt) is 32 lanes x 16 halves.
__device__ __forceinline__ v16bf load_b_packed(const bf16* __restrict__ P, int NT,
                                               int kt, int nt) {
  int lane = threadIdx.x & 31;
  return *(const v16bf*)(P + (((size_t)kt * NT + nt) * 32 + lane) * 16);
}

// B fragment when B^T is stored row-major (Bt[N][K]): one aligned 32B load.
__device__ __forceinline__ v16bf load_b_frag_cm(const bf16* __restrict__ Bt, int ldt,
                                                int k0, int n0) {
  int lane = threadIdx.x & 31;
  return *(const v16bf*)(Bt + (size_t)(n0 + (lane & 15)) * ldt + k0 + ((lane >> 4) << 4));
}

// ---------- weight packing into fragment layout ----------
// dst element idx: e=idx&15, lane=(idx>>4)&31, frag=idx>>9, nt=frag%NT, kt=frag/NT
// value = B[k][n] with k=kt*32+((lane>>4)<<4)+e, n=nt*16+(lane&15)
// B[k][n] = src[k*sk + n*sn + off]  (covers row-major weights and Wc transpose)
__global__ void k_pack_b_f32(const float* __restrict__ src, bf16* __restrict__ dst,
                             int NT, long long sk, long long sn, long long off,
                             int kmax, int total) {
  int idx = blockIdx.x * 256 + threadIdx.x;
  if (idx >= total) return;
  int e  = idx & 15;
  int l  = (idx >> 4) & 31;
  int frag = idx >> 9;
  int nt = frag % NT, kt = frag / NT;
  int k = kt * 32 + ((l >> 4) << 4) + e;
  int n = nt * 16 + (l & 15);
  float v = (k < kmax) ? src[(long long)k * sk + (long long)n * sn + off] : 0.0f;
  dst[idx] = (bf16)v;
}

__global__ void k_pack_b_bf16(const bf16* __restrict__ src, bf16* __restrict__ dst,
                              int NT, int ldb, int kmax, int total) {
  int idx = blockIdx.x * 256 + threadIdx.x;
  if (idx >= total) return;
  int e  = idx & 15;
  int l  = (idx >> 4) & 31;
  int frag = idx >> 9;
  int nt = frag % NT, kt = frag / NT;
  int k = kt * 32 + ((l >> 4) << 4) + e;
  int n = nt * 16 + (l & 15);
  dst[idx] = (k < kmax) ? src[(size_t)k * ldb + n] : (bf16)0.0f;
}

// ---------- gather + gated projection: out = sigmoid(xW1+b1)*tanh(xW2+b2) ----------
__global__ __launch_bounds__(256) void k_proj(const int* __restrict__ ids,
                                              const float* __restrict__ emb,
                                              const bf16* __restrict__ W1p,
                                              const float* __restrict__ b1,
                                              const bf16* __restrict__ W2p,
                                              const float* __restrict__ b2,
                                              bf16* __restrict__ outb) {
  __shared__ __align__(32) bf16 sx[16 * 320];
  __shared__ int sids[16];
  int row0 = blockIdx.x * 16;
  int lane = threadIdx.x & 31, wv = threadIdx.x >> 5;
  if (threadIdx.x < 16) sids[threadIdx.x] = ids[row0 + threadIdx.x];
  __syncthreads();
  for (int r = wv; r < 16; r += 8) {
    const float* erow = emb + (size_t)sids[r] * En;
    for (int k = lane; k < 320; k += 32)
      sx[r * 320 + k] = (k < En) ? (bf16)erow[k] : (bf16)0.0f;
  }
  __syncthreads();
  int hi = lane >> 4;
  int nt0 = wv * 2;
  v8f acc1[2] = {}, acc2[2] = {};
  for (int kt = 0; kt < 10; ++kt) {
    v16bf a = load_a_frag(sx, 320, 0, kt * 32);
#pragma unroll
    for (int j = 0; j < 2; ++j) {
      acc1[j] = wmma_bf16(a, load_b_packed(W1p, 16, kt, nt0 + j), acc1[j]);
      acc2[j] = wmma_bf16(a, load_b_packed(W2p, 16, kt, nt0 + j), acc2[j]);
    }
  }
#pragma unroll
  for (int j = 0; j < 2; ++j) {
    int n = (nt0 + j) * 16 + (lane & 15);
    float bb1 = b1[n], bb2 = b2[n];
#pragma unroll
    for (int i = 0; i < 8; ++i) {
      int r = row0 + hi * 8 + i;
      float x1 = acc1[j][i] + bb1;
      float x2 = acc2[j][i] + bb2;
      float g = (1.0f / (1.0f + expf(-x1))) * tanhf(x2);
      outb[(size_t)r * Mn + n] = (bf16)g;
    }
  }
}

// ---------- attention: att[q][t] = q_p[q] . a_p[t] ----------
__global__ __launch_bounds__(256) void k_att(const bf16* __restrict__ qpb,
                                             const bf16* __restrict__ apb,
                                             float* __restrict__ att) {
  int t0 = (blockIdx.x * 8 + (threadIdx.x >> 5)) * 16;
  int q0 = blockIdx.y * 16;
  v8f acc = {};
  for (int k0 = 0; k0 < Mn; k0 += 32) {
    v16bf a = load_a_frag(qpb, Mn, q0, k0);
    v16bf b = load_b_frag_cm(apb, Mn, k0, t0);   // B = a_p^T, contiguous 32B
    acc = wmma_bf16(a, b, acc);
  }
  int lane = threadIdx.x & 31, hi = lane >> 4, n = lane & 15;
#pragma unroll
  for (int i = 0; i < 8; ++i)
    att[(size_t)(q0 + hi * 8 + i) * Tn + t0 + n] = acc[i];
}

// ---------- per-query softmax stats over T ----------
__global__ __launch_bounds__(256) void k_stats(const float* __restrict__ att,
                                               float* __restrict__ rowmax,
                                               float* __restrict__ rowinv) {
  __shared__ float red[256];
  int q = blockIdx.x;
  const float* row = att + (size_t)q * Tn;
  float m = -3e38f;
  for (int t = threadIdx.x; t < Tn; t += 256) m = fmaxf(m, row[t]);
  red[threadIdx.x] = m; __syncthreads();
  for (int s = 128; s > 0; s >>= 1) {
    if (threadIdx.x < s) red[threadIdx.x] = fmaxf(red[threadIdx.x], red[threadIdx.x + s]);
    __syncthreads();
  }
  m = red[0]; __syncthreads();
  float sum = 0.0f;
  for (int t = threadIdx.x; t < Tn; t += 256) sum += expf(row[t] - m);
  red[threadIdx.x] = sum; __syncthreads();
  for (int s = 128; s > 0; s >>= 1) {
    if (threadIdx.x < s) red[threadIdx.x] += red[threadIdx.x + s];
    __syncthreads();
  }
  if (threadIdx.x == 0) { rowmax[q] = m; rowinv[q] = 1.0f / red[0]; }
}

// ---------- ctx = w_att @ q_p, sim = a_p * ctx (fused) ----------
__global__ __launch_bounds__(256) void k_ctx_sim(const float* __restrict__ att,
                                                 const float* __restrict__ rowmax,
                                                 const float* __restrict__ rowinv,
                                                 const bf16* __restrict__ qpp,
                                                 const bf16* __restrict__ apb,
                                                 bf16* __restrict__ simb) {
  __shared__ float wq[Qn * 16];   // [q][t_local]
  int t0 = blockIdx.x * 16;
  for (int idx = threadIdx.x; idx < Qn * 16; idx += 256) {
    int q = idx >> 4, tl = idx & 15;
    wq[idx] = expf(att[(size_t)q * Tn + t0 + tl] - rowmax[q]) * rowinv[q];
  }
  __syncthreads();
  int lane = threadIdx.x & 31, hi = lane >> 4, r = lane & 15;
  int nt0 = (threadIdx.x >> 5) * 2;
  v8f acc[2] = {};
#pragma unroll
  for (int kt = 0; kt < 2; ++kt) {
    int k0 = kt * 32;
    v16bf a;   // A[r][k] = wq[k][r]  (transposed build from LDS)
#pragma unroll
    for (int j = 0; j < 8; ++j) {
      a[j]     = (bf16)wq[(k0 + hi * 16 + j) * 16 + r];
      a[j + 8] = (bf16)wq[(k0 + hi * 16 + 8 + j) * 16 + r];
    }
#pragma unroll
    for (int j = 0; j < 2; ++j)
      acc[j] = wmma_bf16(a, load_b_packed(qpp, 16, kt, nt0 + j), acc[j]);
  }
#pragma unroll
  for (int j = 0; j < 2; ++j) {
    int n = (nt0 + j) * 16 + r;
#pragma unroll
    for (int i = 0; i < 8; ++i) {
      size_t off = (size_t)(t0 + hi * 8 + i) * Mn + n;
      float simv = acc[j][i] * (float)apb[off];
      simb[off] = (bf16)simv;
    }
  }
}

// ---------- conv (width W) + fused bias/ReLU/segment max-pool, one answer/block ----------
template <int W>
__global__ __launch_bounds__(256) void k_conv_pool(const bf16* __restrict__ simb,
                                                   const bf16* __restrict__ wcp,
                                                   const float* __restrict__ bc,
                                                   bf16* __restrict__ poolsb) {
  __shared__ __align__(32) bf16 s[68 * 256];   // 64 sim rows + 4 zero guard rows
  int k = blockIdx.x;
  for (int idx = threadIdx.x; idx < 68 * 256; idx += 256) {
    int row = idx >> 8, col = idx & 255;
    s[idx] = (row < 64) ? simb[((size_t)k * 64 + row) * Mn + col] : (bf16)0.0f;
  }
  __syncthreads();
  int lane = threadIdx.x & 31, hi = lane >> 4;
  int nt0 = (threadIdx.x >> 5) * 2;
  v8f acc[4][2] = {};
  for (int dt = 0; dt < W; ++dt) {
    const bf16* wp = wcp + (size_t)dt * 65536;   // 8kt * 16nt * 512 per tap
    for (int kt = 0; kt < 8; ++kt) {
      v16bf b0 = load_b_packed(wp, 16, kt, nt0);
      v16bf b1 = load_b_packed(wp, 16, kt, nt0 + 1);
#pragma unroll
      for (int rt = 0; rt < 4; ++rt) {
        v16bf a = load_a_frag(s, 256, rt * 16 + dt, kt * 32);
        acc[rt][0] = wmma_bf16(a, b0, acc[rt][0]);
        acc[rt][1] = wmma_bf16(a, b1, acc[rt][1]);
      }
    }
  }
  const int woff = (W - 1) * 256;
#pragma unroll
  for (int j = 0; j < 2; ++j) {
    int n = (nt0 + j) * 16 + (lane & 15);
    float bcv = bc[n];
    float cmax = -3e38f;
#pragma unroll
    for (int rt = 0; rt < 4; ++rt) {
#pragma unroll
      for (int i = 0; i < 8; ++i) {
        int p = rt * 16 + hi * 8 + i;                 // conv position within answer
        float v = fmaxf(acc[rt][j][i] + bcv, 0.0f);   // ReLU(c + bc)
        if (p > 64 - W) v = -3e38f;                   // crosses answer boundary
        cmax = fmaxf(cmax, v);
      }
    }
    cmax = fmaxf(cmax, __shfl_xor(cmax, 16));
    if (lane < 16)
      poolsb[(size_t)k * 1280 + woff + n] = (bf16)cmax;
  }
}

// ---------- h = tanh(pools @ Wl + bl); logits = h @ Ws + bs ----------
__global__ __launch_bounds__(256) void k_final(const bf16* __restrict__ poolsb,
                                               const bf16* __restrict__ Wlp,
                                               const float* __restrict__ bl,
                                               const float* __restrict__ Ws,
                                               const float* __restrict__ bs,
                                               float* __restrict__ logits) {
  __shared__ float part[16];
  if (threadIdx.x < 16) part[threadIdx.x] = 0.0f;
  __syncthreads();
  int row0 = blockIdx.x * 16;
  int lane = threadIdx.x & 31, hi = lane >> 4;
  int nt0 = (threadIdx.x >> 5) * 2;
  v8f acc[2] = {};
  for (int kt = 0; kt < 40; ++kt) {
    v16bf a = load_a_frag(poolsb, 1280, row0, kt * 32);
#pragma unroll
    for (int j = 0; j < 2; ++j)
      acc[j] = wmma_bf16(a, load_b_packed(Wlp, 16, kt, nt0 + j), acc[j]);
  }
#pragma unroll
  for (int j = 0; j < 2; ++j) {
    int n = (nt0 + j) * 16 + (lane & 15);
    float blv = bl[n], wsv = Ws[n];
#pragma unroll
    for (int i = 0; i < 8; ++i) {
      float h = tanhf(acc[j][i] + blv);
      atomicAdd(&part[hi * 8 + i], h * wsv);
    }
  }
  __syncthreads();
  if (threadIdx.x < 16) logits[row0 + threadIdx.x] = part[threadIdx.x] + bs[0];
}

__global__ __launch_bounds__(1024) void k_logsoftmax(const float* __restrict__ logits,
                                                     float* __restrict__ out) {
  __shared__ float red[1024];
  int i = threadIdx.x;
  float x = logits[i];
  red[i] = x; __syncthreads();
  for (int s = 512; s > 0; s >>= 1) {
    if (i < s) red[i] = fmaxf(red[i], red[i + s]);
    __syncthreads();
  }
  float m = red[0]; __syncthreads();
  red[i] = expf(x - m); __syncthreads();
  for (int s = 512; s > 0; s >>= 1) {
    if (i < s) red[i] += red[i + s];
    __syncthreads();
  }
  out[i] = x - (m + logf(red[0]));
}

extern "C" void kernel_launch(void* const* d_in, const int* in_sizes, int n_in,
                              void* d_out, int out_size, void* d_ws, size_t ws_size,
                              hipStream_t stream) {
  (void)in_sizes; (void)n_in; (void)out_size; (void)ws_size;
  const int*   data_q  = (const int*)d_in[0];
  const int*   data_as = (const int*)d_in[1];
  const float* emb     = (const float*)d_in[3];
  const float* W1      = (const float*)d_in[4];
  const float* b1      = (const float*)d_in[5];
  const float* W2      = (const float*)d_in[6];
  const float* b2      = (const float*)d_in[7];
  const float* Wl      = (const float*)d_in[8];
  const float* bl      = (const float*)d_in[9];
  const float* Ws      = (const float*)d_in[10];
  const float* bs      = (const float*)d_in[11];
  const float* Wc[5]; const float* bc[5];
  for (int i = 0; i < 5; ++i) { Wc[i] = (const float*)d_in[12 + 2 * i];
                                bc[i] = (const float*)d_in[13 + 2 * i]; }

  // workspace carve (256B aligned)
  char* p = (char*)d_ws;
  auto carve = [&](size_t bytes) {
    void* r = (void*)p;
    p += (bytes + 255) & ~(size_t)255;
    return r;
  };
  bf16*  qpb    = (bf16*)carve((size_t)Qn * Mn * 2);
  bf16*  qpp    = (bf16*)carve((size_t)2 * 16 * 512 * 2);         // packed q_p (K=64)
  bf16*  apb    = (bf16*)carve((size_t)Tn * Mn * 2);
  float* att    = (float*)carve((size_t)Qn * Tn * 4);
  float* rowmax = (float*)carve(Qn * 4);
  float* rowinv = (float*)carve(Qn * 4);
  bf16*  simb   = (bf16*)carve((size_t)Tn * Mn * 2);
  bf16*  poolsb = (bf16*)carve((size_t)Kn * 1280 * 2);
  float* logits = (float*)carve(Kn * 4);
  bf16*  W1p    = (bf16*)carve((size_t)10 * 16 * 512 * 2);        // K=320(pad), N=256
  bf16*  W2p    = (bf16*)carve((size_t)10 * 16 * 512 * 2);
  bf16*  Wlp    = (bf16*)carve((size_t)40 * 16 * 512 * 2);        // K=1280, N=256
  bf16*  wcp[5];
  for (int i = 0; i < 5; ++i)
    wcp[i] = (bf16*)carve((size_t)(i + 1) * 8 * 16 * 512 * 2);    // per-tap 8kt*16nt

  // ---- pack weights into fragment layout ----
  // W1/W2: B[k][n] = W[k*256 + n]    (sk=256, sn=1)
  k_pack_b_f32<<<(10 * 16 * 512 + 255) / 256, 256, 0, stream>>>(
      W1, W1p, 16, 256, 1, 0, En, 10 * 16 * 512);
  k_pack_b_f32<<<(10 * 16 * 512 + 255) / 256, 256, 0, stream>>>(
      W2, W2p, 16, 256, 1, 0, En, 10 * 16 * 512);
  // Wl: [1280][256]
  k_pack_b_f32<<<(40 * 16 * 512 + 255) / 256, 256, 0, stream>>>(
      Wl, Wlp, 16, 256, 1, 0, 1280, 40 * 16 * 512);
  // Wc tap d: B[k=i][n=o] = Wc[(o*256+i)*w + d]  (sk=w, sn=256*w, off=d)
  for (int i = 0; i < 5; ++i) {
    int w = i + 1;
    for (int d = 0; d < w; ++d)
      k_pack_b_f32<<<(8 * 16 * 512 + 255) / 256, 256, 0, stream>>>(
          Wc[i], wcp[i] + (size_t)d * 65536, 16, w, 256LL * w, d, 256, 8 * 16 * 512);
  }

  // ---- projections ----
  k_proj<<<Qn / 16, 256, 0, stream>>>(data_q, emb, W1p, b1, W2p, b2, qpb);
  k_proj<<<Tn / 16, 256, 0, stream>>>(data_as, emb, W1p, b1, W2p, b2, apb);
  // pack q_p for the ctx GEMM (B[k=q][n=m] = q_p[q][m], K=64)
  k_pack_b_bf16<<<(2 * 16 * 512 + 255) / 256, 256, 0, stream>>>(
      qpb, qpp, 16, 256, Qn, 2 * 16 * 512);

  // ---- attention + softmax stats + context/sim ----
  k_att<<<dim3(Tn / 128, Qn / 16), 256, 0, stream>>>(qpb, apb, att);
  k_stats<<<Qn, 256, 0, stream>>>(att, rowmax, rowinv);
  k_ctx_sim<<<Tn / 16, 256, 0, stream>>>(att, rowmax, rowinv, qpp, apb, simb);

  // ---- conv + dynamic max pool (one answer per block) ----
  k_conv_pool<1><<<Kn, 256, 0, stream>>>(simb, wcp[0], bc[0], poolsb);
  k_conv_pool<2><<<Kn, 256, 0, stream>>>(simb, wcp[1], bc[1], poolsb);
  k_conv_pool<3><<<Kn, 256, 0, stream>>>(simb, wcp[2], bc[2], poolsb);
  k_conv_pool<4><<<Kn, 256, 0, stream>>>(simb, wcp[3], bc[3], poolsb);
  k_conv_pool<5><<<Kn, 256, 0, stream>>>(simb, wcp[4], bc[4], poolsb);

  // ---- final MLP + scoring + log-softmax ----
  k_final<<<Kn / 16, 256, 0, stream>>>(poolsb, Wlp, bl, Ws, bs, logits);
  k_logsoftmax<<<1, 1024, 0, stream>>>(logits, (float*)d_out);
}
